// TowerRecurrentSBP_23373212025329
// MI455X (gfx1250) — matrix operand order repaired and checked by reading, not compiled, not run on hardware
//
#include <hip/hip_runtime.h>
#include <hip/hip_bf16.h>
#include <math.h>

// ---------------------------------------------------------------------------
// MI455X (gfx1250) TowerRecurrentSBP.
//  - Convs/deconvs: implicit GEMM on v_wmma_f32_16x16x32_f16 (wave32),
//    2x2 register blocking: each wave holds 4 C-tiles (2 M x 2 N), 4 WMMAs
//    per K-chunk sharing 2 A- and 2 B-fragments.
//  - Activations f16 NHWC; B tile staged global->LDS via
//    GLOBAL_LOAD_ASYNC_TO_LDS_B128 (ASYNCcnt + s_wait_asynccnt).
//  - Weights pre-packed once per launch into exact A-fragment VGPR order.
//  - C-tile store: packed b128 of 8 f16 per lane per tile.
// ---------------------------------------------------------------------------

typedef _Float16 half16 __attribute__((ext_vector_type(16)));
typedef _Float16 half8  __attribute__((ext_vector_type(8)));
typedef float    float8 __attribute__((ext_vector_type(8)));

#if defined(__has_builtin)
#  if __has_builtin(__builtin_amdgcn_global_load_async_to_lds_b128)
#    define ASYNC_LDS 1
#  endif
#endif

#ifdef ASYNC_LDS
// builtin signature (probe-confirmed): (int4 AS1*, int4 AS3*, imm offset, imm cpol)
typedef int v4i __attribute__((ext_vector_type(4)));
typedef __attribute__((address_space(1))) v4i g1_v4i;
typedef __attribute__((address_space(3))) v4i l3_v4i;
__device__ __forceinline__ void wait_asynccnt0() {
#  if __has_builtin(__builtin_amdgcn_s_wait_asynccnt)
  __builtin_amdgcn_s_wait_asynccnt(0);
#  else
  asm volatile("s_wait_asynccnt 0x0" ::: "memory");
#  endif
}
#endif

// ---------------- math helpers ----------------
__device__ __forceinline__ float softplusf(float x) {
  return (x > 20.f) ? x : log1pf(expf(x));
}
__device__ __forceinline__ float stdf(float lv) {        // softplus(lv+0.5)+1e-8
  return softplusf(lv + 0.5f) + 1e-8f;
}
__device__ __forceinline__ float sigf(float x) {
  return 1.f / (1.f + expf(-x));
}
__device__ __forceinline__ float logsigf(float x) {      // log(sigmoid(x)), stable
  return (x >= 0.f) ? -log1pf(expf(-x)) : (x - log1pf(expf(x)));
}

// ---------------------------------------------------------------------------
// Weight packing: W[Cout,Cin,kh,kw] f32 -> Wp f16 in A-fragment order.
// Tile ((mt*taps + t)*nCb + cb): 512 f16 = [lane][elem]:
//   m  = mt*16 + (lane&15)
//   ic = cb*32 + ((lane>=16)?16:0) + e
// ---------------------------------------------------------------------------
__global__ void pack_weights(const float* __restrict__ W, _Float16* __restrict__ Wp,
                             int Cout, int Cin, int taps, long total) {
  const long i = (long)blockIdx.x * blockDim.x + threadIdx.x;
  if (i >= total) return;
  const int nCb  = (Cin + 31) >> 5;
  const long tile = i >> 9;
  const int  j    = (int)(i & 511);
  const int  L    = j >> 4;
  const int  e    = j & 15;
  const int  cb   = (int)(tile % nCb);
  const long t2   = tile / nCb;
  const int  t    = (int)(t2 % taps);
  const int  mt   = (int)(t2 / taps);
  const int  m    = mt * 16 + (L & 15);
  const int  ic   = cb * 32 + ((L >= 16) ? 16 : 0) + e;
  float v = 0.f;
  if (m < Cout && ic < Cin)
    v = W[((long)m * Cin + ic) * taps + t];
  Wp[i] = (_Float16)v;
}

// ---------------------------------------------------------------------------
// C-tile store helper: f16 NHWC, packed 8xf16 when Cout allows.
// ---------------------------------------------------------------------------
__device__ __forceinline__ void store_tile(float8 acc, _Float16* __restrict__ y,
                                           const float* __restrict__ bias,
                                           int Cout, int mBase, long pix,
                                           int relu, int accumulate) {
  _Float16* dst = y + pix * Cout + mBase;
  if ((Cout & 7) == 0) {
    half8 prev = {};
    if (accumulate) prev = *(const half8*)dst;
    half8 o;
#pragma unroll
    for (int r = 0; r < 8; ++r) {
      float v = acc[r];
      if (accumulate) v += (float)prev[r];
      if (bias)       v += bias[mBase + r];
      if (relu)       v = v > 0.f ? v : 0.f;
      o[r] = (_Float16)v;
    }
    *(half8*)dst = o;
  } else {
#pragma unroll
    for (int r = 0; r < 8; ++r) {
      const int m = mBase + r;
      if (m < Cout) {
        float v = acc[r];
        if (accumulate) v += (float)dst[r];
        if (bias)       v += bias[m];
        if (relu)       v = v > 0.f ? v : 0.f;
        dst[r] = (_Float16)v;
      }
    }
  }
}

// ---------------------------------------------------------------------------
// Implicit-GEMM conv / deconv, f16 NHWC activations, 2x2 blocking.
//   Block: 32-pixel tile (2 N-subtiles); wave owns 2 M-tiles -> 4 accumulators.
//   K loop: taps x 32-channel blocks; 4 WMMAs per chunk per wave.
// mode==0: conv (stride,pad).  mode==1: transposed conv stride 2.
// srcF16==1: xA/xB f16 NHWC (async fast path; Cin, CinA multiples of 32).
// srcF16==0: xA f32 NCHW (stem1 only).
// ---------------------------------------------------------------------------
__global__ __launch_bounds__(256)
void conv_wmma(const void* __restrict__ xAv, const void* __restrict__ xBv,
               int CinA, int srcF16,
               const _Float16* __restrict__ Wp, const float* __restrict__ bias,
               _Float16* __restrict__ y,
               int N, int Cin, int Hin, int Win,
               int Cout, int Hout, int Wout,
               int kh, int kw, int stride, int pad,
               int mode, int relu, int accumulate, int pxTiles) {
  __shared__ __align__(32) _Float16 ldsB[1024];          // 2 N-subtiles
  __shared__ int ptab_b[32], ptab_oy[32], ptab_ox[32], ptab_v[32];

  const int waveId = threadIdx.x >> 5;
  const int lane   = threadIdx.x & 31;
  const int lh     = lane & 15;
  const int hihalf = lane >> 4;

  const int pxT = blockIdx.x % pxTiles;
  const int ocB = blockIdx.x / pxTiles;
  const int wpb = blockDim.x >> 5;                       // waves per block
  const int ocTiles = (Cout + 15) >> 4;
  const int mt0raw = ocB * (2 * wpb) + waveId;
  const int mt1raw = ocB * (2 * wpb) + wpb + waveId;
  const int mt0 = (mt0raw < ocTiles) ? mt0raw : (ocTiles - 1);
  const int mt1 = (mt1raw < ocTiles) ? mt1raw : (ocTiles - 1);
  const bool st0 = (mt0raw < ocTiles);
  const bool st1 = (mt1raw < ocTiles);

  const long Npix = (long)N * Hout * Wout;
  if (threadIdx.x < 32) {
    const long n = (long)pxT * 32 + threadIdx.x;
    int v = (n < Npix) ? 1 : 0;
    int b = 0, oy = 0, ox = 0;
    if (v) {
      b = (int)(n / (Hout * Wout));
      const int r = (int)(n % (Hout * Wout));
      oy = r / Wout; ox = r % Wout;
    }
    ptab_b[threadIdx.x] = b; ptab_oy[threadIdx.x] = oy;
    ptab_ox[threadIdx.x] = ox; ptab_v[threadIdx.x] = v;
  }

  const int taps = kh * kw;
  const int nCb  = (Cin + 31) >> 5;
  const int CinB = Cin - CinA;
  const _Float16* Wp0 = Wp + (long)mt0 * taps * nCb * 512;
  const _Float16* Wp1 = Wp + (long)mt1 * taps * nCb * 512;

  float8 acc00 = {}, acc01 = {}, acc10 = {}, acc11 = {};
  for (int t = 0; t < taps; ++t) {
    const int ky = t / kw, kx = t % kw;
    for (int cb = 0; cb < nCb; ++cb) {
      __syncthreads();                       // prior chunk's reads done
      if (srcF16) {
        // ---- fast path: stage both N-subtiles, f16 NHWC, async b128 ----
        for (int sr = threadIdx.x; sr < 64; sr += blockDim.x) {
          const int L   = sr & 31;           // fragment lane-row
          const int sub = sr >> 5;           // N-subtile 0/1
          const int n   = sub * 16 + (L & 15);
          const int cOff = cb * 32 + ((L >= 16) ? 16 : 0);
          int b = ptab_b[n], oy = ptab_oy[n], ox = ptab_ox[n];
          int iy = 0, ix = 0; bool ok = (ptab_v[n] != 0);
          if (ok) {
            if (mode == 0) {
              iy = oy * stride + ky - pad;
              ix = ox * stride + kx - pad;
              ok = (iy >= 0 && iy < Hin && ix >= 0 && ix < Win);
            } else {
              const int ty = oy - ky + pad;
              const int tx = ox - kx + pad;
              ok = false;
              if (((ty | tx) >= 0) && (((ty | tx) & 1) == 0)) {
                iy = ty >> 1; ix = tx >> 1;
                ok = (iy < Hin && ix < Win);
              }
            }
          }
          _Float16* l = &ldsB[sub * 512 + L * 16];
          if (ok) {
            const _Float16* src; int Crow, c0;
            if (cOff < CinA) { src = (const _Float16*)xAv; Crow = CinA; c0 = cOff; }
            else             { src = (const _Float16*)xBv; Crow = CinB; c0 = cOff - CinA; }
            const _Float16* g = src + ((long)(b * Hin + iy) * Win + ix) * Crow + c0;
#ifdef ASYNC_LDS
            __builtin_amdgcn_global_load_async_to_lds_b128(
                (g1_v4i*)g,       (l3_v4i*)l,       0, 0);
            __builtin_amdgcn_global_load_async_to_lds_b128(
                (g1_v4i*)(g + 8), (l3_v4i*)(l + 8), 0, 0);
#else
            *(half8*)l       = *(const half8*)g;
            *(half8*)(l + 8) = *(const half8*)(g + 8);
#endif
          } else {
            const half16 zz = {};
            *(half16*)l = zz;
          }
        }
#ifdef ASYNC_LDS
        wait_asynccnt0();
#endif
      } else {
        // ---- slow path (stem1 only): cooperative f32 NCHW gather ----
        for (int i = threadIdx.x; i < 1024; i += blockDim.x) {
          const int sub = i >> 9;
          const int j   = i & 511;
          const int L = j >> 4, e = j & 15;
          const int n = sub * 16 + (L & 15);
          const int ic = cb * 32 + ((L >= 16) ? 16 : 0) + e;
          _Float16 v = (_Float16)0.f;
          if (ic < Cin && ptab_v[n]) {
            const int b = ptab_b[n], oy = ptab_oy[n], ox = ptab_ox[n];
            const int iy = oy * stride + ky - pad;
            const int ix = ox * stride + kx - pad;
            if (iy >= 0 && iy < Hin && ix >= 0 && ix < Win) {
              const float* xf = (const float*)xAv;
              v = (_Float16)xf[(((long)b * Cin + ic) * Hin + iy) * Win + ix];
            }
          }
          ldsB[i] = v;
        }
      }
      __syncthreads();
      const half16 bf0 = *(const half16*)&ldsB[lane * 16];
      const half16 bf1 = *(const half16*)&ldsB[512 + lane * 16];
      const long  toff = ((long)(t * nCb + cb)) * 512 + lane * 16;
      const half16 af0 = *(const half16*)&Wp0[toff];
      const half16 af1 = *(const half16*)&Wp1[toff];
      acc00 = __builtin_amdgcn_wmma_f32_16x16x32_f16(false, af0, false, bf0,
                                                     (short)0, acc00, false, false);
      acc01 = __builtin_amdgcn_wmma_f32_16x16x32_f16(false, af0, false, bf1,
                                                     (short)0, acc01, false, false);
      acc10 = __builtin_amdgcn_wmma_f32_16x16x32_f16(false, af1, false, bf0,
                                                     (short)0, acc10, false, false);
      acc11 = __builtin_amdgcn_wmma_f32_16x16x32_f16(false, af1, false, bf1,
                                                     (short)0, acc11, false, false);
    }
  }

  // ---- store 4 C-tiles: columns from ptab, rows mt*16 + hihalf*8 .. +7 ----
  const int mb0 = mt0 * 16 + hihalf * 8;
  const int mb1 = mt1 * 16 + hihalf * 8;
  // N-subtile 0 column
  if (ptab_v[lh]) {
    const long pix0 = (long)(ptab_b[lh] * Hout + ptab_oy[lh]) * Wout + ptab_ox[lh];
    if (st0) store_tile(acc00, y, bias, Cout, mb0, pix0, relu, accumulate);
    if (st1) store_tile(acc10, y, bias, Cout, mb1, pix0, relu, accumulate);
  }
  // N-subtile 1 column
  if (ptab_v[16 + lh]) {
    const long pix1 = (long)(ptab_b[16 + lh] * Hout + ptab_oy[16 + lh]) * Wout + ptab_ox[16 + lh];
    if (st0) store_tile(acc01, y, bias, Cout, mb0, pix1, relu, accumulate);
    if (st1) store_tile(acc11, y, bias, Cout, mb1, pix1, relu, accumulate);
  }
}

// ---------------------------------------------------------------------------
// Elementwise stages. Activations f16 NHWC; LSTM cell state f32; outputs f32.
// ---------------------------------------------------------------------------
#define BQ 32
#define ZD 32
#define HP 1024          // 32*32 latent plane
#define KST 6

// g: f16 NHWC [B,32,32,64]; eps0: f32 NCHW; z: f16 NHWC [B,32,32,32]
__global__ void gauss_sample_kl(const _Float16* __restrict__ g,
                                const float* __restrict__ eps0,
                                _Float16* __restrict__ z,
                                float* __restrict__ klout) {
  const long i = (long)blockIdx.x * blockDim.x + threadIdx.x;
  const long tot = (long)BQ * HP * ZD;
  if (i >= tot) return;
  const int  zc  = (int)(i & (ZD - 1));
  const long row = i >> 5;                  // global pixel index
  const int  b   = (int)(row >> 10);
  const int  p   = (int)(row & (HP - 1));
  const float mu = (float)g[row * 64 + zc];
  const float lv = (float)g[row * 64 + 32 + zc];
  const float s  = stdf(lv);
  z[row * 32 + zc] = (_Float16)(mu + s * eps0[(((long)b * ZD + zc) << 10) + p]);
  const float kl = -logf(s) + 0.5f * (s * s + mu * mu) - 0.5f;
  klout[((((long)b * ZD + zc) << 10) + p) * KST + 0] = kl;
}

// gates: f16 NHWC [B,32,32,256] (i,f,g,o x 64ch); c: f32 NHWC; h: f16 NHWC
__global__ void lstm_pointwise(const _Float16* __restrict__ gates,
                               float* __restrict__ c, _Float16* __restrict__ h) {
  const long i = (long)blockIdx.x * blockDim.x + threadIdx.x;
  const long tot = (long)BQ * HP * 64;
  if (i >= tot) return;
  const int  ch  = (int)(i & 63);
  const long row = i >> 6;
  const long base = row * 256;
  const float gi = (float)gates[base + ch];
  const float gf = (float)gates[base + 64  + ch];
  const float gg = (float)gates[base + 128 + ch];
  const float go = (float)gates[base + 192 + ch];
  const float c2 = sigf(gf) * c[i] + sigf(gi) * tanhf(gg);
  c[i] = c2;
  h[i] = (_Float16)(sigf(go) * tanhf(c2));
}

__global__ void qp_sample_kl(const _Float16* __restrict__ qs,
                             const _Float16* __restrict__ ps,
                             const float* __restrict__ epsk,
                             _Float16* __restrict__ z,
                             float* __restrict__ klout, int kslot) {
  const long i = (long)blockIdx.x * blockDim.x + threadIdx.x;
  const long tot = (long)BQ * HP * ZD;
  if (i >= tot) return;
  const int  zc  = (int)(i & (ZD - 1));
  const long row = i >> 5;
  const int  b   = (int)(row >> 10);
  const int  p   = (int)(row & (HP - 1));
  const float qmu = (float)qs[row * 64 + zc];
  const float qlv = (float)qs[row * 64 + 32 + zc];
  const float pmu = (float)ps[row * 64 + zc];
  const float plv = (float)ps[row * 64 + 32 + zc];
  const float qsd = stdf(qlv);
  const float psd = stdf(plv);
  z[row * 32 + zc] = (_Float16)(qmu + qsd * epsk[(((long)b * ZD + zc) << 10) + p]);
  const float d  = qmu - pmu;
  const float kl = logf(psd / qsd) + (qsd * qsd + d * d) / (2.f * psd * psd) - 0.5f;
  klout[((((long)b * ZD + zc) << 10) + p) * KST + kslot] = kl;
}

__global__ void stick_break(const _Float16* __restrict__ deck,
                            float* __restrict__ logss,
                            float* __restrict__ out, int last) {
  const long i = (long)blockIdx.x * blockDim.x + threadIdx.x;
  const long tot = (long)BQ * 128 * 128;
  if (i >= tot) return;
  const float d = (float)deck[i];
  const float ls = logss[i];
  out[i] = ls + logsigf(d);
  if (!last) logss[i] = ls + logsigf(-d);
}

// ---------------------------------------------------------------------------
// Host side
// ---------------------------------------------------------------------------
static inline long packed_size(int Cout, int Cin, int taps) {
  return (long)((Cout + 15) / 16) * taps * ((Cin + 31) / 32) * 512;
}

static inline void launch_conv(hipStream_t s,
                               const void* xA, const void* xB, int CinA, int srcF16,
                               const _Float16* Wp, const float* bias, _Float16* y,
                               int N, int Cin, int Hin, int Win,
                               int Cout, int Hout, int Wout,
                               int kh, int kw, int stride, int pad,
                               int mode, int relu, int accum) {
  const long npix    = (long)N * Hout * Wout;
  const int  pxTiles = (int)((npix + 31) / 32);           // 32-pixel N tile
  const int  ocTiles = (Cout + 15) / 16;
  int wpb = (ocTiles + 1) / 2;                            // 2 M-tiles per wave
  if (wpb > 8) wpb = 8;
  const int  ocBlocks = (ocTiles + 2 * wpb - 1) / (2 * wpb);
  conv_wmma<<<pxTiles * ocBlocks, 32 * wpb, 0, s>>>(
      xA, xB, CinA, srcF16, Wp, bias, y,
      N, Cin, Hin, Win, Cout, Hout, Wout,
      kh, kw, stride, pad, mode, relu, accum, pxTiles);
}

extern "C" void kernel_launch(void* const* d_in, const int* in_sizes, int n_in,
                              void* d_out, int out_size, void* d_ws, size_t ws_size,
                              hipStream_t stream) {
  (void)in_sizes; (void)n_in; (void)out_size; (void)ws_size;
  const float* x         = (const float*)d_in[0];
  const float* eps       = (const float*)d_in[1];
  const float* stem1_W   = (const float*)d_in[2];
  const float* stem1_b   = (const float*)d_in[3];
  const float* stem2_W   = (const float*)d_in[4];
  const float* stem2_b   = (const float*)d_in[5];
  const float* gauss_W   = (const float*)d_in[6];
  const float* gauss_b   = (const float*)d_in[7];
  const float* post_Wx   = (const float*)d_in[8];
  const float* post_Wh   = (const float*)d_in[9];
  const float* post_b    = (const float*)d_in[10];
  const float* postlin_W = (const float*)d_in[11];
  const float* postlin_b = (const float*)d_in[12];
  const float* prior_Wx  = (const float*)d_in[13];
  const float* prior_Wh  = (const float*)d_in[14];
  const float* prior_b   = (const float*)d_in[15];
  const float* priorlin_W= (const float*)d_in[16];
  const float* priorlin_b= (const float*)d_in[17];
  const float* dec1_W    = (const float*)d_in[18];
  const float* dec1_b    = (const float*)d_in[19];
  const float* dec2_W    = (const float*)d_in[20];
  const float* dec2_b    = (const float*)d_in[21];

  float* out = (float*)d_out;

  // ---------------- workspace bump allocator (64B aligned) ----------------
  char* cur = (char*)d_ws;
  auto alloc = [&](long bytes) -> void* {
    void* p = (void*)cur;
    cur += (bytes + 63) & ~63L;
    return p;
  };
  // f16 NHWC activations
  _Float16* h1f   = (_Float16*)alloc(32L * 64 * 64 * 128 * 2);  // stem1 / dec1 out
  _Float16* h2f   = (_Float16*)alloc(32L * 32 * 32 * 128 * 2);
  _Float16* gqf   = (_Float16*)alloc(32L * 32 * 32 * 64 * 2);   // gauss / qstats
  _Float16* psf   = (_Float16*)alloc(32L * 32 * 32 * 64 * 2);
  _Float16* zsf   = (_Float16*)alloc(6L * 32 * 32 * 32 * 32 * 2);
  _Float16* hqf   = (_Float16*)alloc(32L * 32 * 32 * 64 * 2);
  _Float16* hpf   = (_Float16*)alloc(32L * 32 * 32 * 64 * 2);
  _Float16* gatesf= (_Float16*)alloc(32L * 32 * 32 * 256 * 2);
  _Float16* deckf = (_Float16*)alloc(32L * 128 * 128 * 2);
  // f32 state
  float* cq    = (float*)alloc(32L * 32 * 32 * 64 * 4);
  float* cp    = (float*)alloc(32L * 32 * 32 * 64 * 4);
  float* logss = (float*)alloc(32L * 128 * 128 * 4);
  // packed weights (f16)
  auto do_pack = [&](const float* W, int Cout, int Cin, int taps) -> _Float16* {
    const long sz = packed_size(Cout, Cin, taps);
    _Float16* p = (_Float16*)alloc(sz * 2);
    pack_weights<<<(int)((sz + 255) / 256), 256, 0, stream>>>(W, p, Cout, Cin, taps, sz);
    return p;
  };
  _Float16* P_stem1   = do_pack(stem1_W,   128, 3,   25);
  _Float16* P_stem2   = do_pack(stem2_W,   128, 128, 25);
  _Float16* P_gauss   = do_pack(gauss_W,   64,  128, 1);
  _Float16* P_postWx  = do_pack(post_Wx,   256, 160, 25);
  _Float16* P_postWh  = do_pack(post_Wh,   256, 64,  25);
  _Float16* P_postlin = do_pack(postlin_W, 64,  64,  1);
  _Float16* P_priorWx = do_pack(prior_Wx,  256, 32,  25);
  _Float16* P_priorWh = do_pack(prior_Wh,  256, 64,  25);
  _Float16* P_priorlin= do_pack(priorlin_W,64,  64,  1);
  _Float16* P_dec1    = do_pack(dec1_W,    128, 32,  16);
  _Float16* P_dec2    = do_pack(dec2_W,    1,   128, 16);

  float* logms_out = out;                                   // [K,B,1,128,128]
  float* kl_out    = out + (long)KST * 32 * 128 * 128;      // [B,ZD,32,32,K]

  const long SZ_Zf = 32L * 32 * 32 * 32;                    // one z slice (f16 elems)
  (void)hipMemsetAsync(cq,    0, 32L * 32 * 32 * 64 * 4, stream);
  (void)hipMemsetAsync(cp,    0, 32L * 32 * 32 * 64 * 4, stream);
  (void)hipMemsetAsync(hqf,   0, 32L * 32 * 32 * 64 * 2, stream);
  (void)hipMemsetAsync(hpf,   0, 32L * 32 * 32 * 64 * 2, stream);
  (void)hipMemsetAsync(logss, 0, 32L * 128 * 128 * 4, stream);

  // ---- encoder stems (5x5 stride 2, SAME pad=1, ReLU) ----
  launch_conv(stream, x,   nullptr, 3,   0, P_stem1, stem1_b, h1f,
              32, 3,   128, 128, 128, 64, 64, 5, 5, 2, 1, 0, 1, 0);
  launch_conv(stream, h1f, nullptr, 128, 1, P_stem2, stem2_b, h2f,
              32, 128, 64,  64,  128, 32, 32, 5, 5, 2, 1, 0, 1, 0);

  // ---- gaussian head (1x1) + first sample / KL(N(0,1)) ----
  launch_conv(stream, h2f, nullptr, 128, 1, P_gauss, gauss_b, gqf,
              32, 128, 32, 32, 64, 32, 32, 1, 1, 1, 0, 0, 0, 0);
  {
    const long tot = (long)BQ * HP * ZD;
    gauss_sample_kl<<<(int)((tot + 255) / 256), 256, 0, stream>>>(gqf, eps, zsf, kl_out);
  }

  // ---- recurrent posterior/prior, 5 steps ----
  for (int sidx = 0; sidx < KST - 1; ++sidx) {
    _Float16* zprev = zsf + (long)sidx * SZ_Zf;
    _Float16* zcur  = zsf + (long)(sidx + 1) * SZ_Zf;
    const float* epsk = eps + (long)(sidx + 1) * SZ_Zf;

    launch_conv(stream, h2f, zprev, 128, 1, P_postWx, post_b, gatesf,
                32, 160, 32, 32, 256, 32, 32, 5, 5, 1, 2, 0, 0, 0);
    launch_conv(stream, hqf, nullptr, 64, 1, P_postWh, nullptr, gatesf,
                32, 64, 32, 32, 256, 32, 32, 5, 5, 1, 2, 0, 0, 1);
    {
      const long tot = (long)BQ * HP * 64;
      lstm_pointwise<<<(int)((tot + 255) / 256), 256, 0, stream>>>(gatesf, cq, hqf);
    }
    launch_conv(stream, hqf, nullptr, 64, 1, P_postlin, postlin_b, gqf,
                32, 64, 32, 32, 64, 32, 32, 1, 1, 1, 0, 0, 0, 0);

    launch_conv(stream, zprev, nullptr, 32, 1, P_priorWx, prior_b, gatesf,
                32, 32, 32, 32, 256, 32, 32, 5, 5, 1, 2, 0, 0, 0);
    launch_conv(stream, hpf, nullptr, 64, 1, P_priorWh, nullptr, gatesf,
                32, 64, 32, 32, 256, 32, 32, 5, 5, 1, 2, 0, 0, 1);
    {
      const long tot = (long)BQ * HP * 64;
      lstm_pointwise<<<(int)((tot + 255) / 256), 256, 0, stream>>>(gatesf, cp, hpf);
    }
    launch_conv(stream, hpf, nullptr, 64, 1, P_priorlin, priorlin_b, psf,
                32, 64, 32, 32, 64, 32, 32, 1, 1, 1, 0, 0, 0, 0);

    {
      const long tot = (long)BQ * HP * ZD;
      qp_sample_kl<<<(int)((tot + 255) / 256), 256, 0, stream>>>(
          gqf, psf, epsk, zcur, kl_out, sidx + 1);
    }
  }

  // ---- decoder + stick-breaking, per latent step ----
  for (int k = 0; k < KST; ++k) {
    _Float16* zk = zsf + (long)k * SZ_Zf;
    launch_conv(stream, zk, nullptr, 32, 1, P_dec1, dec1_b, h1f,
                32, 32, 32, 32, 128, 64, 64, 4, 4, 2, 1, 1, 1, 0);
    launch_conv(stream, h1f, nullptr, 128, 1, P_dec2, dec2_b, deckf,
                32, 128, 64, 64, 1, 128, 128, 4, 4, 2, 1, 1, 0, 0);
    {
      const long tot = (long)BQ * 128 * 128;
      stick_break<<<(int)((tot + 255) / 256), 256, 0, stream>>>(
          deckf, logss, logms_out + (long)k * tot, (k == KST - 1) ? 1 : 0);
    }
  }
}
